// MobileBertSelfAttention_83992380440911
// MI455X (gfx1250) — matrix operand order repaired and checked by reading, not compiled, not run on hardware
//
#include <hip/hip_runtime.h>
#include <hip/hip_bf16.h>

// MobileBert single-head self-attention, flash-attention style, CDNA5 WMMA.
// proj: q = (Xq@Wq+bq)*log2e/sqrt(128) (f16), k = Xk@Wk+bk (f16),
//       vT = (Xv@Wv+bv)^T (f16, [b][d][s])
// fa  : online-softmax (base-2) flash attention, f16 WMMA, f32 accumulation,
//       double-buffered K/V^T tiles staged with async global->LDS copies.

typedef __attribute__((ext_vector_type(16))) _Float16 v16h;
typedef __attribute__((ext_vector_type(8)))  float    v8f;
typedef __attribute__((ext_vector_type(4)))  unsigned int v4u;
typedef __attribute__((ext_vector_type(4)))  int      v4i;

#define WMMA_F16(A, B, C) \
  __builtin_amdgcn_wmma_f32_16x16x32_f16(false, (A), false, (B), (short)0, (C), false, false)

// ---- async global->LDS copy (CDNA5 ASYNCcnt path), with safe fallback ------
#if defined(__gfx1250__) && __has_builtin(__builtin_amdgcn_global_load_async_to_lds_b128)
#define ASYNC_LDS 1
#else
#define ASYNC_LDS 0
#endif

#if ASYNC_LDS
typedef __attribute__((address_space(1))) v4i as1_v4i;
typedef __attribute__((address_space(3))) v4i as3_v4i;
#endif

// 16B async copy; OFFB is a compile-time byte offset applied to BOTH the
// global and LDS address (matches the ISA's shared INST_OFFSET semantics).
template <int OFFB>
__device__ inline void cp16o(_Float16* dst_lds, const _Float16* src_glb) {
#if ASYNC_LDS
  __builtin_amdgcn_global_load_async_to_lds_b128(
      (as1_v4i*)src_glb, (as3_v4i*)dst_lds, OFFB, 0);
#else
  *(v4u*)(dst_lds + OFFB / 2) = *(const v4u*)(src_glb + OFFB / 2);
#endif
}

__device__ inline void async_wait_all() {
#if ASYNC_LDS
#if __has_builtin(__builtin_amdgcn_s_wait_asynccnt)
  __builtin_amdgcn_s_wait_asynccnt(0);
#else
  asm volatile("s_wait_asynccnt 0x0" ::: "memory");
#endif
#endif
}

union HF16 { v16h h; struct { v4u lo; v4u hi; } q; };
union PK8 { _Float16 h[8]; v4u u4; };

// A-fragment (16x32 f16, M x K), per ISA table:
// lanes 0-15: M=lane, VGPR0-3 -> K=0..7, VGPR4-7 -> K=16..23
// lanes 16-31: M=lane-16, VGPR0-3 -> K=8..15, VGPR4-7 -> K=24..31
__device__ inline v16h frag_ld_a16(const _Float16* base, int ld, int kbase, int lane) {
  const int g = (lane >> 4) & 1, m = lane & 15;
  const _Float16* row = base + (size_t)m * ld + kbase + g * 8;
  HF16 f;
  f.q.lo = *(const v4u*)(row);
  f.q.hi = *(const v4u*)(row + 16);
  return f.h;
}

// Same A layout, converting from f32 source.
__device__ inline v16h frag_ld_a32(const float* base, int ld, int kbase, int lane) {
  const int g = (lane >> 4) & 1, m = lane & 15;
  const float* row = base + (size_t)m * ld + kbase + g * 8;
  v16h h;
#pragma unroll
  for (int i = 0; i < 8; ++i) h[i] = (_Float16)row[i];
#pragma unroll
  for (int i = 0; i < 8; ++i) h[8 + i] = (_Float16)row[16 + i];
  return h;
}

// B-fragment (32x16 f16, K x N): lanes 0-15: N=lane, K=0..7,16..23;
// lanes 16-31: N=lane-16, K=8..15,24..31. Source transposed row-major matT[n][k].
__device__ inline v16h frag_ld_b16(const _Float16* matT, int ld, int kbase, int lane) {
  const int g = (lane >> 4) & 1, n = lane & 15;
  const _Float16* row = matT + (size_t)n * ld + kbase + g * 8;
  HF16 f;
  f.q.lo = *(const v4u*)(row);
  f.q.hi = *(const v4u*)(row + 16);
  return f.h;
}

// ---------------------------------------------------------------------------
// Projection: out[row][n] = (sum_k X[row][k]*W[k][n] + bias[n]) * outscale
// ---------------------------------------------------------------------------
__global__ __launch_bounds__(128) void proj_kernel(
    const float* __restrict__ X, const float* __restrict__ W,
    const float* __restrict__ bias, _Float16* __restrict__ out,
    int din, int transpose_out, float outscale)
{
  __shared__ __align__(16) _Float16 sWt[128][40];  // Wt[n][k] for current k-chunk

  const int tid  = threadIdx.x;
  const int wave = tid >> 5;
  const int lane = tid & 31;
  const int g    = (lane >> 4) & 1;
  const int m    = lane & 15;
  const int row0 = (blockIdx.x * 4 + wave) * 16;
  const float* xb = X + (size_t)row0 * din;

  v8f acc[8] = {};

  for (int kc = 0; kc < din; kc += 32) {
    __syncthreads();
    // thread tid owns output column n = tid: 32 coalesced b32 reads of W rows,
    // packed into 4 b128 LDS stores (transpose done in registers).
    {
      const float* wcol = W + (size_t)kc * 128 + tid;
#pragma unroll
      for (int k8 = 0; k8 < 4; ++k8) {
        PK8 pk;
#pragma unroll
        for (int kk = 0; kk < 8; ++kk)
          pk.h[kk] = (_Float16)wcol[(size_t)(k8 * 8 + kk) * 128];
        *(v4u*)&sWt[tid][k8 * 8] = pk.u4;
      }
    }
    __syncthreads();

    const v16h af = frag_ld_a32(xb, din, kc, lane);
#pragma unroll
    for (int j = 0; j < 8; ++j) {
      const v16h bf = frag_ld_b16(&sWt[j * 16][0], 40, 0, lane);
      acc[j] = WMMA_F16(af, bf, acc[j]);
    }
  }

#pragma unroll
  for (int j = 0; j < 8; ++j) {
    const float bj = bias[j * 16 + m];
#pragma unroll
    for (int r = 0; r < 8; ++r) {
      const _Float16 val = (_Float16)((acc[j][r] + bj) * outscale);
      const int row = row0 + r + 8 * g;
      const int col = j * 16 + m;
      if (transpose_out) {
        const int bb = row >> 12, s = row & 4095;
        out[((size_t)bb * 128 + col) * 4096 + s] = val;
      } else {
        out[(size_t)row * 128 + col] = val;
      }
    }
  }
}

// ---------------------------------------------------------------------------
// Flash attention: 1 block = 64 query rows (4 waves x 16), key blocks of 64,
// double-buffered async-staged tiles.
// ---------------------------------------------------------------------------
__global__ __launch_bounds__(128) void fa_kernel(
    const _Float16* __restrict__ Q, const _Float16* __restrict__ K,
    const _Float16* __restrict__ VT, const float* __restrict__ mask,
    float* __restrict__ out)
{
  __shared__ __align__(16) _Float16 sK[2][64][128];    // K tiles [key][d]
  __shared__ __align__(16) _Float16 sVT[2][128][64];   // V^T tiles [d][key]
  __shared__ __align__(16) _Float16 sP[4][16][64];     // per-wave P [qrow][key]

  const int b    = blockIdx.y;
  const int tid  = threadIdx.x;
  const int wave = tid >> 5;
  const int lane = tid & 31;
  const int g    = (lane >> 4) & 1;
  const int m    = lane & 15;
  const int qrow0 = blockIdx.x * 64 + wave * 16;

  const _Float16* Kb  = K  + (size_t)b * 4096 * 128;
  const _Float16* VTb = VT + (size_t)b * 128 * 4096;

  // per-thread staging bases, hoisted out of the key loop
  _Float16* lK0 = &sK[0][0][0] + tid * 8;           // K LDS: chunk u at +u*2048B
  _Float16* lK1 = &sK[1][0][0] + tid * 8;
  _Float16* lV0 = &sVT[0][0][0] + tid * 8;          // V^T LDS: chunk u at +u*2048B
  _Float16* lV1 = &sVT[1][0][0] + tid * 8;
  const _Float16* gV = VTb + (size_t)(tid >> 3) * 4096 + (tid & 7) * 8;

  // stage tiles for key block kb into buffer buf (async if available)
  auto stage = [&](int buf, int kb) {
    // K tile: global & LDS advance in lockstep -> shared immediate offsets
    const _Float16* gk = Kb + (size_t)kb * 128 + tid * 8;
    _Float16* lk = buf ? lK1 : lK0;
    cp16o<0>(lk, gk);      cp16o<2048>(lk, gk);
    cp16o<4096>(lk, gk);   cp16o<6144>(lk, gk);
    cp16o<8192>(lk, gk);   cp16o<10240>(lk, gk);
    cp16o<12288>(lk, gk);  cp16o<14336>(lk, gk);
    // V^T tile: strides differ (LDS +2048B, global +16 rows), step pointers
    const _Float16* gv = gV + kb;
    _Float16* lv = buf ? lV1 : lV0;
#pragma unroll
    for (int u = 0; u < 8; ++u) {
      cp16o<0>(lv, gv);
      lv += 1024;             // 2048 B
      gv += (size_t)16 * 4096;
    }
  };

  // Q fragments resident in registers (q already scaled by log2e/sqrt(128))
  const _Float16* qbase = Q + ((size_t)b * 4096 + qrow0) * 128;
  v16h qf[4];
#pragma unroll
  for (int c = 0; c < 4; ++c) qf[c] = frag_ld_a16(qbase, 128, 32 * c, lane);

  v8f o[8] = {};
  float mstat[8], lstat[8];
#pragma unroll
  for (int r = 0; r < 8; ++r) { mstat[r] = -1e30f; lstat[r] = 0.0f; }

  const float log2e = 1.4426950408889634f;
  const float* mrow = mask + (size_t)b * 4096;

  stage(0, 0);
  async_wait_all();
  __syncthreads();

  for (int it = 0; it < 64; ++it) {
    const int kb  = it * 64;
    const int cur = it & 1;
    if (it + 1 < 64) stage(cur ^ 1, kb + 64);  // prefetch next tiles

    const _Float16(*cK)[128] = sK[cur];
    const _Float16(*cVT)[64] = sVT[cur];

    // T = Q' K^T + mask*log2e (log2-domain logits); A operand fixed per c-run
    v8f sacc[4] = {};
#pragma unroll
    for (int c = 0; c < 4; ++c)
#pragma unroll
      for (int j = 0; j < 4; ++j) {
        const v16h kf = frag_ld_b16(&cK[j * 16][0], 128, 32 * c, lane);
        sacc[j] = WMMA_F16(qf[c], kf, sacc[j]);
      }
#pragma unroll
    for (int j = 0; j < 4; ++j) {
      const float mval = mrow[kb + j * 16 + m] * log2e;
#pragma unroll
      for (int r = 0; r < 8; ++r) sacc[j][r] += mval;
    }

    // row max across j-tiles, then across the fragment's 16-lane group
    float rmax[8];
#pragma unroll
    for (int r = 0; r < 8; ++r)
      rmax[r] = fmaxf(fmaxf(sacc[0][r], sacc[1][r]), fmaxf(sacc[2][r], sacc[3][r]));
#pragma unroll
    for (int off = 1; off < 16; off <<= 1)
#pragma unroll
      for (int r = 0; r < 8; ++r)
        rmax[r] = fmaxf(rmax[r], __shfl_xor(rmax[r], off, 32));

    float alpha[8], rsum[8];
#pragma unroll
    for (int r = 0; r < 8; ++r) {
      const float mn = fmaxf(mstat[r], rmax[r]);
      alpha[r] = exp2f(mstat[r] - mn);
      mstat[r] = mn;
      rsum[r] = 0.0f;
    }

    // P = exp2(T - m), transpose C-layout -> A-layout via per-wave LDS tile
    _Float16* pb = &sP[wave][0][0];
#pragma unroll
    for (int j = 0; j < 4; ++j)
#pragma unroll
      for (int r = 0; r < 8; ++r) {
        const float p = exp2f(sacc[j][r] - mstat[r]);
        rsum[r] += p;
        pb[(size_t)(r + 8 * g) * 64 + j * 16 + m] = (_Float16)p;
      }
#pragma unroll
    for (int off = 1; off < 16; off <<= 1)
#pragma unroll
      for (int r = 0; r < 8; ++r) rsum[r] += __shfl_xor(rsum[r], off, 32);
#pragma unroll
    for (int r = 0; r < 8; ++r) lstat[r] = lstat[r] * alpha[r] + rsum[r];
#pragma unroll
    for (int t = 0; t < 8; ++t)
#pragma unroll
      for (int r = 0; r < 8; ++r) o[t][r] *= alpha[r];

    // O += P V (8 d-tiles x 2 k-chunks); A operand fixed per c-run
    const v16h pf[2] = { frag_ld_a16(pb, 64, 0, lane),
                         frag_ld_a16(pb, 64, 32, lane) };
#pragma unroll
    for (int c = 0; c < 2; ++c)
#pragma unroll
      for (int t = 0; t < 8; ++t) {
        const v16h vf = frag_ld_b16(&cVT[t * 16][0], 64, 32 * c, lane);
        o[t] = WMMA_F16(pf[c], vf, o[t]);
      }

    async_wait_all();   // prefetched tiles landed in LDS (this wave)
    __syncthreads();    // visible to all waves; safe to overwrite cur next iter
  }

  float inv[8];
#pragma unroll
  for (int r = 0; r < 8; ++r) inv[r] = 1.0f / lstat[r];
  float* obase = out + ((size_t)b * 4096 + qrow0) * 128;
#pragma unroll
  for (int t = 0; t < 8; ++t)
#pragma unroll
    for (int r = 0; r < 8; ++r)
      obase[(size_t)(r + 8 * g) * 128 + t * 16 + m] = o[t][r] * inv[r];
}

extern "C" void kernel_launch(void* const* d_in, const int* in_sizes, int n_in,
                              void* d_out, int out_size, void* d_ws, size_t ws_size,
                              hipStream_t stream) {
  const float* q_in = (const float*)d_in[0];
  const float* k_in = (const float*)d_in[1];
  const float* v_in = (const float*)d_in[2];
  const float* msk  = (const float*)d_in[3];
  const float* Wq   = (const float*)d_in[4];
  const float* bq   = (const float*)d_in[5];
  const float* Wk   = (const float*)d_in[6];
  const float* bk   = (const float*)d_in[7];
  const float* Wv   = (const float*)d_in[8];
  const float* bv   = (const float*)d_in[9];
  float* out = (float*)d_out;

  const size_t NR = (size_t)8 * 4096;
  _Float16* qh = (_Float16*)d_ws;      // [B*S][128] f16 (pre-scaled)
  _Float16* kh = qh + NR * 128;        // [B*S][128] f16
  _Float16* vt = kh + NR * 128;        // [B][128][S] f16 (transposed)

  const float qscale = 0.08838834764831845f * 1.4426950408889634f; // log2e/sqrt(128)
  proj_kernel<<<512, 128, 0, stream>>>(q_in, Wq, bq, qh, 128, 0, qscale);
  proj_kernel<<<512, 128, 0, stream>>>(k_in, Wk, bk, kh, 128, 0, 1.0f);
  proj_kernel<<<512, 128, 0, stream>>>(v_in, Wv, bv, vt, 512, 1, 1.0f);
  fa_kernel<<<dim3(64, 8), 128, 0, stream>>>(qh, kh, vt, msk, out);
}
